// AnomalyAttention_48301202210891
// MI455X (gfx1250) — matrix operand order repaired and checked by reading, not compiled, not run on hardware
//
#include <hip/hip_runtime.h>

typedef __attribute__((ext_vector_type(16))) _Float16     v16h;
typedef __attribute__((ext_vector_type(8)))  float        v8f;
typedef __attribute__((ext_vector_type(4)))  unsigned int v4u;
typedef __attribute__((ext_vector_type(8)))  int          v8i;
typedef __attribute__((ext_vector_type(4)))  int          v4i;

#define B_    4
#define L_    2048
#define H_    8
#define E_    64
#define WAVES 8
#define QTILE 128   // query rows per workgroup (8 waves x 16 rows)
#define KTILE 32    // key rows per tile (= WMMA K for f16)

// LDS -> v16h load: halves p[0..7] and p[16..23] (2x ds_load_b128, table k order)
static __device__ inline v16h lds_ld16h(const _Float16* p) {
  v16h r; __builtin_memcpy(&r, p, 16); __builtin_memcpy((_Float16*)&r + 8, p + 16, 16); return r;
}

// ---------------------------------------------------------------------------
// TDM: 2D f16 tile load, tile == tensor (no OOB), data_size = 2 bytes.
// D# packing per CDNA5 ISA 8.3/8.4: group0 = {count|..., lds_addr,
// global_addr[31:0], global_addr[56:32]|type=2}; group1 holds dims/strides.
// amdgpu-toolchain builtin arity: (v4u, v8i, v4i, v4i, v8i, cpol)
// ---------------------------------------------------------------------------
static __device__ inline void tdm_load_2d_f16(unsigned lds_addr,
                                              const _Float16* gptr,
                                              unsigned tile_d0,   // elems, contiguous
                                              unsigned tile_d1,   // rows
                                              unsigned long long d0_stride) // elems
{
  const unsigned long long ga = (unsigned long long)(uintptr_t)gptr;
  v4u g0;
  g0.x = 1u;                                                  // count=1 (valid D#)
  g0.y = lds_addr;                                            // LDS byte address
  g0.z = (unsigned)ga;                                        // global_addr[31:0]
  g0.w = (unsigned)((ga >> 32) & 0x1FFFFFFull) | (2u << 30);  // addr[56:32] | type=2

  v8i g1;
  g1[0] = (int)(1u << 16);                                    // data_size = 1 (2 bytes)
  g1[1] = (int)((tile_d0 & 0xFFFFu) << 16);                   // tensor_dim0[15:0]
  g1[2] = (int)((tile_d0 >> 16) | ((tile_d1 & 0xFFFFu) << 16)); // tdim0 hi | tdim1 lo
  g1[3] = (int)((tile_d1 >> 16) | (tile_d0 << 16));           // tdim1 hi | tile_dim0
  g1[4] = (int)(tile_d1 & 0xFFFFu);                           // tile_dim1 | tile_dim2=0
  g1[5] = (int)(unsigned)(d0_stride & 0xFFFFFFFFull);         // dim0_stride[31:0]
  g1[6] = (int)(unsigned)((d0_stride >> 32) & 0xFFFFull);     // dim0_stride[47:32]
  g1[7] = 0;
  v4i g2 = {0, 0, 0, 0};                                      // 2D: groups 2/3 unused
  v4i g3 = {0, 0, 0, 0};
  v8i g4 = {0, 0, 0, 0, 0, 0, 0, 0};
  __builtin_amdgcn_tensor_load_to_lds(g0, g1, g2, g3, g4, 0);
}

// ---------------------------------------------------------------------------
// Pre-pass: K (f32 [B,L,H,E]) -> f16 [BH][L][E];  V -> f16 transposed [BH][E][L]
// ---------------------------------------------------------------------------
__global__ __launch_bounds__(256)
void cvt_kv_kernel(const float* __restrict__ K, const float* __restrict__ V,
                   _Float16* __restrict__ K16, _Float16* __restrict__ V16t)
{
  const int n = blockIdx.x * 256 + threadIdx.x;    // source-linear (coalesced reads)
  const int e = n & (E_ - 1);
  const int h = (n >> 6) & (H_ - 1);               // E = 64
  const int l = (n >> 9) & (L_ - 1);               // H*E = 512
  const int b = n >> 20;                           // L*H*E = 2^20
  const int bh = b * H_ + h;
  K16[((size_t)bh * L_ + l) * E_ + e]  = (_Float16)K[n];
  V16t[((size_t)bh * E_ + e) * L_ + l] = (_Float16)V[n];
}

// ---------------------------------------------------------------------------
// Flash attention, causal, WMMA f16 -> f32, TDM-staged K/V tiles
// ---------------------------------------------------------------------------
__global__ __launch_bounds__(256, 1)
void fa_causal_wmma_tdm(const float* __restrict__ Q,
                        const _Float16* __restrict__ K16,
                        const _Float16* __restrict__ V16t,
                        float* __restrict__ O)
{
  const int h  = blockIdx.y % H_;
  const int b  = blockIdx.y / H_;
  const int bh = blockIdx.y;
  const int q0 = blockIdx.x * QTILE;

  const int tid  = threadIdx.x;
  const int wave = tid >> 5;
  const int lane = tid & 31;
  const int ln16 = lane & 15;
  const int hi   = lane >> 4;

  __shared__ __align__(32) _Float16 Ksh[2][KTILE][64];   // [buf][key][e]
  __shared__ __align__(32) _Float16 Vsh[2][64][KTILE];   // [buf][e][key]
  __shared__ __align__(32) float    Ssh[WAVES][16][KTILE]; // wave-private score spill

  const size_t strideL = (size_t)H_ * E_;
  const size_t baseBH  = ((size_t)b * L_) * strideL + (size_t)h * E_;

  // ----- Q slab (16x64) -> A-matrix layout, softmax scale folded in (x 0.125) -----
  const int qrow = q0 + wave * 16 + ln16;
  const float* qp = Q + baseBH + (size_t)qrow * strideL;
  v16h qa[2];
  #pragma unroll
  for (int c = 0; c < 2; ++c) {
    _Float16 tmp[16];
    #pragma unroll
    for (int hh = 0; hh < 16; ++hh) {
      const int e = 32 * c + (hh < 8 ? hh : hh + 8) + hi * 8;
      tmp[hh] = (_Float16)(qp[e] * 0.125f);
    }
    __builtin_memcpy(&qa[c], tmp, 32);
  }

  v8f o0 = {}, o1 = {}, o2 = {}, o3 = {};
  float m_run = -1e30f, l_run = 0.0f;      // row-major softmax state (row = lane&15)

  const int qbase_s  = __builtin_amdgcn_readfirstlane(q0 + wave * 16); // scalar
  const int rowmax_s = qbase_s + 15;
  const int nkb      = (q0 + QTILE) / KTILE;

  const _Float16* kbase = K16  + (size_t)bh * L_ * E_;
  const _Float16* vbase = V16t + (size_t)bh * E_ * L_;

  // Each wave DMAs its own slice: K rows 4w..4w+3, V rows (e) 8w..8w+7.
  // ----- prologue: tile 0 -> buffer 0 -----
  {
    const unsigned kl = (unsigned)(uintptr_t)(void*)&Ksh[0][4 * wave][0];
    const unsigned vl = (unsigned)(uintptr_t)(void*)&Vsh[0][8 * wave][0];
    tdm_load_2d_f16(kl, kbase + (size_t)(4 * wave) * E_,      64, 4, 64);
    tdm_load_2d_f16(vl, vbase + (size_t)(8 * wave) * L_,      32, 8, L_);
  }

  for (int kbI = 0; kbI < nkb; ++kbI) {
    const int kb  = kbI * KTILE;
    const int cur = kbI & 1;

    if (kbI + 1 < nkb) {        // issue next tile into the other buffer, then
      const int kn = kb + KTILE; // wait for current tile (2 newest still in flight)
      const unsigned kl = (unsigned)(uintptr_t)(void*)&Ksh[cur ^ 1][4 * wave][0];
      const unsigned vl = (unsigned)(uintptr_t)(void*)&Vsh[cur ^ 1][8 * wave][0];
      tdm_load_2d_f16(kl, kbase + (size_t)(kn + 4 * wave) * E_, 64, 4, 64);
      tdm_load_2d_f16(vl, vbase + (size_t)(8 * wave) * L_ + kn, 32, 8, L_);
      __builtin_amdgcn_s_wait_tensorcnt(2);
    } else {
      __builtin_amdgcn_s_wait_tensorcnt(0);
    }
    __syncthreads();            // every wave's slice of tile kbI landed

    if (kb <= rowmax_s) {       // scalar branch: EXEC stays all-ones for WMMA
      // ---- S = Q K^T : two 16x16 tiles, E=64 as 2 x (K=32) ----
      v8f s[2];
      #pragma unroll
      for (int t = 0; t < 2; ++t) {
        const v16h b0 = lds_ld16h(&Ksh[cur][ln16 + 16 * t][     hi * 8]);
        const v16h b1 = lds_ld16h(&Ksh[cur][ln16 + 16 * t][32 + hi * 8]);
        v8f c = {};
        c = __builtin_amdgcn_wmma_f32_16x16x32_f16(false, qa[0], false, b0, (short)0, c, false, false);
        c = __builtin_amdgcn_wmma_f32_16x16x32_f16(false, qa[1], false, b1, (short)0, c, false, false);
        s[t] = c;
      }

      // ---- spill D-layout scores, re-read row-major (lane = row) ----
      #pragma unroll
      for (int t = 0; t < 2; ++t)
        #pragma unroll
        for (int r = 0; r < 8; ++r)
          Ssh[wave][r + hi * 8][ln16 + 16 * t] = s[t][r];

      float a0[8], a1[8];   // this lane's 16 of the row's 32 cols (A interleave)
      __builtin_memcpy(a0, &Ssh[wave][ln16][     hi * 8], 32);
      __builtin_memcpy(a1, &Ssh[wave][ln16][16 + hi * 8], 32);

      if (kb + KTILE - 1 > qbase_s) {   // diagonal block: causal mask (scalar branch)
        #pragma unroll
        for (int j = 0; j < 8; ++j) {
          if (kb +      hi * 8 + j > qbase_s + ln16) a0[j] = -1e30f;
          if (kb + 16 + hi * 8 + j > qbase_s + ln16) a1[j] = -1e30f;
        }
      }

      // ---- online softmax: per-lane chains + one xor16 to merge row halves ----
      float mx = a0[0];
      #pragma unroll
      for (int j = 1; j < 8; ++j) mx = fmaxf(mx, a0[j]);
      #pragma unroll
      for (int j = 0; j < 8; ++j) mx = fmaxf(mx, a1[j]);
      mx = fmaxf(mx, __shfl_xor(mx, 16, 32));
      const float mnew = fmaxf(m_run, mx);

      _Float16 ph[16];
      float rs = 0.0f;
      #pragma unroll
      for (int j = 0; j < 8; ++j) {
        const float p = __expf(a0[j] - mnew);
        rs += p; ph[j] = (_Float16)p;
      }
      #pragma unroll
      for (int j = 0; j < 8; ++j) {
        const float p = __expf(a1[j] - mnew);
        rs += p; ph[8 + j] = (_Float16)p;
      }
      rs += __shfl_xor(rs, 16, 32);

      const float alpha = __expf(m_run - mnew);
      m_run = mnew;
      l_run = l_run * alpha + rs;

      v16h pa;               // already in A-matrix interleave
      __builtin_memcpy(&pa, ph, 32);

      // ---- rescale O (broadcast alpha into D layout: VGPR r -> row r+hi*8) ----
      #pragma unroll
      for (int r = 0; r < 8; ++r) {
        const float aD = __shfl(alpha, r + hi * 8, 32);
        o0[r] *= aD; o1[r] *= aD; o2[r] *= aD; o3[r] *= aD;
      }

      // ---- O += P * V : 4 tiles, K=32 covers the key block ----
      v16h bv;
      bv = lds_ld16h(&Vsh[cur][ 0 + ln16][hi * 8]);
      o0 = __builtin_amdgcn_wmma_f32_16x16x32_f16(false, pa, false, bv, (short)0, o0, false, false);
      bv = lds_ld16h(&Vsh[cur][16 + ln16][hi * 8]);
      o1 = __builtin_amdgcn_wmma_f32_16x16x32_f16(false, pa, false, bv, (short)0, o1, false, false);
      bv = lds_ld16h(&Vsh[cur][32 + ln16][hi * 8]);
      o2 = __builtin_amdgcn_wmma_f32_16x16x32_f16(false, pa, false, bv, (short)0, o2, false, false);
      bv = lds_ld16h(&Vsh[cur][48 + ln16][hi * 8]);
      o3 = __builtin_amdgcn_wmma_f32_16x16x32_f16(false, pa, false, bv, (short)0, o3, false, false);
    }
    __syncthreads();          // tile consumed; buf[cur] may be overwritten next iter
  }

  // ---- normalize + store (each half-wave writes 64B contiguous) ----
  const float inv = 1.0f / l_run;
  #pragma unroll
  for (int r = 0; r < 8; ++r) {
    const float iD  = __shfl(inv, r + hi * 8, 32);
    const int   row = qbase_s + r + hi * 8;
    float* op = O + baseBH + (size_t)row * strideL;
    op[ 0 + ln16] = o0[r] * iD;
    op[16 + ln16] = o1[r] * iD;
    op[32 + ln16] = o2[r] * iD;
    op[48 + ln16] = o3[r] * iD;
  }
}

extern "C" void kernel_launch(void* const* d_in, const int* in_sizes, int n_in,
                              void* d_out, int out_size, void* d_ws, size_t ws_size,
                              hipStream_t stream) {
  (void)in_sizes; (void)n_in; (void)out_size; (void)ws_size;
  const float* Q = (const float*)d_in[0];
  const float* K = (const float*)d_in[1];
  const float* V = (const float*)d_in[2];
  float* Out = (float*)d_out;

  // workspace: f16 K [BH][L][E] + f16 V^T [BH][E][L]  (16.8 MB total)
  _Float16* K16  = (_Float16*)d_ws;
  _Float16* V16t = K16 + (size_t)B_ * H_ * L_ * E_;

  const int nElem = B_ * L_ * H_ * E_;
  cvt_kv_kernel<<<nElem / 256, 256, 0, stream>>>(K, V, K16, V16t);

  dim3 grid(L_ / QTILE, B_ * H_);
  fa_causal_wmma_tdm<<<grid, dim3(256), 0, stream>>>(Q, K16, V16t, Out);
}
